// RiemannRotaryAttention_63934883168366
// MI455X (gfx1250) — matrix operand order repaired and checked by reading, not compiled, not run on hardware
//
#include <hip/hip_runtime.h>
#include <hip/hip_bf16.h>
#include <math.h>

typedef __attribute__((ext_vector_type(16))) __bf16 v16bf;
typedef __attribute__((ext_vector_type(8)))  float  v8f;

#define B_   2
#define N_   2048
#define D_   512
#define H_   8
#define DH_  64
#define KP_  32
#define TAU_F 6.2831853071795864769f

// gamma[p] = zeros[p] / zeros[0], folded at compile time in fp32
__device__ __constant__ float c_gamma[KP_] = {
  14.134725142f/14.134725142f, 21.022039639f/14.134725142f, 25.01085758f/14.134725142f,
  30.424876126f/14.134725142f, 32.935061588f/14.134725142f, 37.586178159f/14.134725142f,
  40.918719012f/14.134725142f, 43.327073281f/14.134725142f, 48.005150881f/14.134725142f,
  49.773832478f/14.134725142f, 52.970321478f/14.134725142f, 56.446247697f/14.134725142f,
  59.347044003f/14.134725142f, 60.831778525f/14.134725142f, 65.112544048f/14.134725142f,
  67.079810529f/14.134725142f, 69.546401711f/14.134725142f, 72.067157674f/14.134725142f,
  75.704690699f/14.134725142f, 77.144840069f/14.134725142f, 79.33737502f/14.134725142f,
  82.910380854f/14.134725142f, 84.735492981f/14.134725142f, 87.425274613f/14.134725142f,
  88.809111208f/14.134725142f, 92.491899271f/14.134725142f, 94.651344041f/14.134725142f,
  95.870634228f/14.134725142f, 98.831194218f/14.134725142f, 101.317851006f/14.134725142f,
  103.72553804f/14.134725142f, 105.446623052f/14.134725142f
};

__device__ __forceinline__ unsigned short f2bf(float f) {
  unsigned int u = __float_as_uint(f);
  unsigned int r = u + 0x7FFFu + ((u >> 16) & 1u);
  return (unsigned short)(r >> 16);
}

// ---- gfx1250 async memory->LDS copy (ASYNCcnt-tracked DMA path) ----------
// LDS address = low 32 bits of the flat shared pointer.
// 16 bytes per lane per issue.
__device__ __forceinline__ void async_b128(void* lds_ptr, const void* gptr) {
  unsigned lds_off = (unsigned)(unsigned long long)lds_ptr;
  unsigned long long ga = (unsigned long long)gptr;
  asm volatile("global_load_async_to_lds_b128 %0, %1, off"
               :: "v"(lds_off), "v"(ga) : "memory");
}
__device__ __forceinline__ void wait_async0() {
  asm volatile("s_wait_asynccnt 0x0" ::: "memory");
}

// A-fragment (16x32 bf16, MxK), per ISA 7.12.2: lanes 0-15 -> M=row, K 0..7 & 16..23;
// lanes 16-31 -> K 8..15 & 24..31. Two 16B LDS loads per lane.
__device__ __forceinline__ v16bf load_a_frag(const unsigned short* lds, int stride,
                                             int row0, int koff) {
  const int lane = threadIdx.x & 31;
  const int half = lane >> 4;
  const unsigned short* p = lds + (row0 + (lane & 15)) * stride + koff + half * 8;
  union { v16bf v; uint4 q[2]; } u;
  u.q[0] = *(const uint4*)p;
  u.q[1] = *(const uint4*)(p + 16);
  return u.v;
}

// B-fragment (32x16 bf16, KxN): lane -> column; lanes 0-15 K 0..7 & 16..23, lanes 16-31 K 8..15 & 24..31.
// Source LDS layout is [col][K-contiguous], so two 16B LDS loads per lane.
__device__ __forceinline__ v16bf load_b_frag(const unsigned short* lds, int stride,
                                             int col0, int koff) {
  const int lane = threadIdx.x & 31;
  const int half = lane >> 4;
  const unsigned short* p = lds + (col0 + (lane & 15)) * stride + koff + half * 8;
  union { v16bf v; uint4 q[2]; } u;
  u.q[0] = *(const uint4*)p;
  u.q[1] = *(const uint4*)(p + 16);
  return u.v;
}

__global__ void k_f32_to_bf16(const float* __restrict__ in, unsigned short* __restrict__ o, int n) {
  int i = blockIdx.x * blockDim.x + threadIdx.x;
  if (i < n) o[i] = f2bf(in[i]);
}

// cos/sin tables (h,n,32) + per-key bias  kbias[h,t] = lam_h * sum_p floor(theta/tau) / 32
// (the sheet bias is rank-1 under the causal mask since sheet is monotone in t;
//  the per-row part cancels in softmax, leaving only this per-key term)
__global__ void k_tables(const float* __restrict__ log_scale,
                         const float* __restrict__ log_lambda,
                         float* __restrict__ cosT, float* __restrict__ sinT,
                         float* __restrict__ kbias) {
  int id = blockIdx.x * blockDim.x + threadIdx.x;
  if (id >= H_ * N_) return;
  int h = id / N_, t = id - h * N_;
  float sc  = __expf(log_scale[h]);
  float lam = __expf(log_lambda[h]);
  float lt  = log1pf((float)t);
  float acc = 0.f;
  #pragma unroll
  for (int p = 0; p < KP_; ++p) {
    float th = lt * c_gamma[p] * sc;
    cosT[(size_t)id * KP_ + p] = cosf(th);
    sinT[(size_t)id * KP_ + p] = sinf(th);
    acc += floorf(th / TAU_F);
  }
  kbias[id] = lam * acc * (1.0f / (float)KP_);
}

// C[MxNc] f32 = A[MxK] bf16 * B[KxNc] bf16. Block tile 128x64, 8 waves -> 32x32 each.
// All leading dims / K are template constants so addressing folds to immediates.
template <int LDA, int LDB, int LDC, int KDIM>
__global__ __launch_bounds__(256) void k_gemm_bf16(
    const unsigned short* __restrict__ A,
    const unsigned short* __restrict__ Bm,
    float* __restrict__ C) {
  __shared__ __align__(16) unsigned short sA[128 * 32];
  __shared__ __align__(16) unsigned short sB[64 * 32];   // transposed: [col][k]
  const int tid  = threadIdx.x;
  const int lane = tid & 31, wave = tid >> 5;
  const int half = lane >> 4, l15 = lane & 15;
  const int wm = (wave & 3) * 32, wn = (wave >> 2) * 32;
  const int m0 = blockIdx.y * 128, n0 = blockIdx.x * 64;

  v8f z = {};
  v8f acc[2][2];
  acc[0][0] = z; acc[0][1] = z; acc[1][0] = z; acc[1][1] = z;

  for (int kb = 0; kb < KDIM; kb += 32) {
    __syncthreads();
    // A tile 128x32 via async DMA to LDS (2 x b128 per thread)
    #pragma unroll
    for (int i = 0; i < 2; ++i) {
      int chunk = i * 256 + tid;                 // 512 chunks of 8 elems
      int r = chunk >> 2, k8 = (chunk & 3) << 3;
      async_b128(sA + r * 32 + k8, A + (size_t)(m0 + r) * LDA + kb + k8);
    }
    // B tile 32x64 -> LDS transposed (manual: needs transpose)
    #pragma unroll
    for (int i = 0; i < 8; ++i) {
      int idx = i * 256 + tid;                   // 2048 elems
      int k = idx >> 6, c = idx & 63;
      sB[c * 32 + k] = Bm[(size_t)(kb + k) * LDB + n0 + c];
    }
    wait_async0();
    __syncthreads();
    #pragma unroll
    for (int mi = 0; mi < 2; ++mi) {
      v16bf a = load_a_frag(sA, 32, wm + mi * 16, 0);
      #pragma unroll
      for (int ni = 0; ni < 2; ++ni) {
        v16bf b = load_b_frag(sB, 32, wn + ni * 16, 0);
        acc[mi][ni] = __builtin_amdgcn_wmma_f32_16x16x32_bf16(
            false, a, false, b, (short)0, acc[mi][ni], false, false);
      }
    }
  }
  float* cptr = C + (size_t)(m0 + wm + half * 8) * LDC + n0 + wn + l15;
  #pragma unroll
  for (int mi = 0; mi < 2; ++mi)
    #pragma unroll
    for (int ni = 0; ni < 2; ++ni)
      #pragma unroll
      for (int j = 0; j < 8; ++j)
        cptr[(mi * 16 + j) * LDC + ni * 16] = acc[mi][ni][j];
}

// rotary on q/k, scatter qkv (B,N,3,H,dh) f32 -> q/k/v bf16 in [B,H,N,dh]
__global__ void k_rotary(const float* __restrict__ qkvf,
                         const float* __restrict__ cosT, const float* __restrict__ sinT,
                         unsigned short* __restrict__ qbf, unsigned short* __restrict__ kbf,
                         unsigned short* __restrict__ vbf) {
  int id = blockIdx.x * blockDim.x + threadIdx.x;   // B*H*N*KP = 2^20
  int p = id & 31;
  int n = (id >> 5) & (N_ - 1);
  int h = (id >> 16) & 7;
  int b = id >> 19;
  float c = cosT[((size_t)h * N_ + n) * KP_ + p];
  float s = sinT[((size_t)h * N_ + n) * KP_ + p];
  const size_t row = (size_t)(b * N_ + n) * (3 * D_);
  const int col = h * DH_ + 2 * p;
  float q1 = qkvf[row + col],           q2 = qkvf[row + col + 1];
  float k1 = qkvf[row + D_ + col],      k2 = qkvf[row + D_ + col + 1];
  float v1 = qkvf[row + 2 * D_ + col],  v2 = qkvf[row + 2 * D_ + col + 1];
  size_t o = ((size_t)(b * H_ + h) * N_ + n) * DH_ + 2 * p;
  qbf[o]     = f2bf(q1 * c - q2 * s);
  qbf[o + 1] = f2bf(q1 * s + q2 * c);
  kbf[o]     = f2bf(k1 * c - k2 * s);
  kbf[o + 1] = f2bf(k1 * s + k2 * c);
  vbf[o]     = f2bf(v1);
  vbf[o + 1] = f2bf(v2);
}

// Flash attention: block = (b,h,qtile of 128 rows); 8 waves x 16 query rows; 64-key tiles.
__global__ __launch_bounds__(256) void k_attn(
    const unsigned short* __restrict__ qbf,
    const unsigned short* __restrict__ kbf,
    const unsigned short* __restrict__ vbf,
    const float* __restrict__ kbias,
    unsigned short* __restrict__ obf) {
  __shared__ __align__(16) unsigned short sQ[128 * 64];
  __shared__ __align__(16) unsigned short sK[64 * 64];
  __shared__ __align__(16) unsigned short sVt[64 * 64];   // transposed: [d][key]
  __shared__ __align__(16) unsigned short sP[8 * 16 * 64];
  __shared__ float sKb[64];
  const int tid  = threadIdx.x;
  const int lane = tid & 31, wave = tid >> 5;
  const int half = lane >> 4, l15 = lane & 15;
  const int qt = blockIdx.x;
  const int bh = blockIdx.y;
  const int b = bh >> 3;
  const int h = bh & 7;
  const int q0 = qt * 128;

  // Q tile 128x64 via async DMA to LDS (4 x b128 per thread)
  const unsigned short* qg = qbf + ((size_t)bh * N_ + q0) * DH_;
  #pragma unroll
  for (int i = 0; i < 4; ++i) {
    int chunk = i * 256 + tid;                   // 1024 chunks of 8 elems
    async_b128(sQ + chunk * 8, qg + chunk * 8);
  }

  v8f z = {};
  v8f Oacc[4];
  float m_i[8], l_i[8];
  #pragma unroll
  for (int t = 0; t < 4; ++t) Oacc[t] = z;
  #pragma unroll
  for (int j = 0; j < 8; ++j) { m_i[j] = -1e30f; l_i[j] = 0.f; }

  const int nk = 2 * (qt + 1);
  const unsigned short* kg0 = kbf + (size_t)bh * N_ * DH_;
  const unsigned short* vg0 = vbf + (size_t)bh * N_ * DH_;
  const float* kbh = kbias + h * N_;

  for (int kt = 0; kt < nk; ++kt) {
    const int j0 = kt * 64;
    __syncthreads();
    const unsigned short* kg = kg0 + (size_t)j0 * DH_;
    const unsigned short* vg = vg0 + (size_t)j0 * DH_;
    // K tile via async DMA (2 x b128 per thread)
    #pragma unroll
    for (int i = 0; i < 2; ++i) {
      int chunk = i * 256 + tid;                 // 512 chunks of 8 elems
      async_b128(sK + chunk * 8, kg + chunk * 8);
    }
    // V tile transposed into LDS: vectorized 8B global reads, scalar LDS stores
    #pragma unroll
    for (int i = 0; i < 4; ++i) {
      int chunk = i * 256 + tid;                 // 1024 chunks of 4 elems
      int r = chunk >> 4, d0 = (chunk & 15) << 2;
      unsigned long long pack = *(const unsigned long long*)(vg + (size_t)r * DH_ + d0);
      unsigned short e0 = (unsigned short)pack, e1 = (unsigned short)(pack >> 16);
      unsigned short e2 = (unsigned short)(pack >> 32), e3 = (unsigned short)(pack >> 48);
      sVt[(d0 + 0) * 64 + r] = e0;
      sVt[(d0 + 1) * 64 + r] = e1;
      sVt[(d0 + 2) * 64 + r] = e2;
      sVt[(d0 + 3) * 64 + r] = e3;
    }
    if (tid < 64) sKb[tid] = kbh[j0 + tid];
    // prefetch next K/V tiles into cache while we compute this one
    if (kt + 1 < nk) {
      __builtin_prefetch(kg + (size_t)64 * DH_ + (tid & 63) * 64, 0, 0);
      __builtin_prefetch(vg + (size_t)64 * DH_ + (tid & 63) * 64, 0, 0);
    }
    wait_async0();
    __syncthreads();

    v8f Sacc[4];
    #pragma unroll
    for (int t = 0; t < 4; ++t) Sacc[t] = z;
    #pragma unroll
    for (int ks = 0; ks < 2; ++ks) {            // S = Qr * Kr^T over dh=64
      v16bf a = load_a_frag(sQ, 64, wave * 16, ks * 32);
      #pragma unroll
      for (int t = 0; t < 4; ++t) {
        v16bf bf = load_b_frag(sK, 64, t * 16, ks * 32);
        Sacc[t] = __builtin_amdgcn_wmma_f32_16x16x32_bf16(
            false, a, false, bf, (short)0, Sacc[t], false, false);
      }
    }

    const int grow_base = q0 + wave * 16 + half * 8;
    #pragma unroll
    for (int j = 0; j < 8; ++j) {
      const int grow = grow_base + j;
      float rm = -1e30f;
      #pragma unroll
      for (int t = 0; t < 4; ++t) {
        int gcol = j0 + t * 16 + l15;
        float s = Sacc[t][j] * 0.125f + sKb[t * 16 + l15];
        if (gcol > grow) s = -1e30f;            // causal mask
        Sacc[t][j] = s;
        rm = fmaxf(rm, s);
      }
      #pragma unroll
      for (int off = 1; off < 16; off <<= 1) rm = fmaxf(rm, __shfl_xor(rm, off, 32));
      float mnew = fmaxf(m_i[j], rm);
      float fac = __expf(m_i[j] - mnew);
      m_i[j] = mnew;
      float ps = 0.f;
      #pragma unroll
      for (int t = 0; t < 4; ++t) {
        float p = __expf(Sacc[t][j] - mnew);
        Sacc[t][j] = p;
        ps += p;
      }
      #pragma unroll
      for (int off = 1; off < 16; off <<= 1) ps += __shfl_xor(ps, off, 32);
      l_i[j] = l_i[j] * fac + ps;
      #pragma unroll
      for (int t = 0; t < 4; ++t) Oacc[t][j] *= fac;
    }

    // stage P (wave-private LDS) into A-fragment layout, then O += P * V
    unsigned short* sPw = sP + wave * (16 * 64);
    #pragma unroll
    for (int j = 0; j < 8; ++j)
      #pragma unroll
      for (int t = 0; t < 4; ++t)
        sPw[(j + half * 8) * 64 + t * 16 + l15] = f2bf(Sacc[t][j]);

    #pragma unroll
    for (int ks = 0; ks < 2; ++ks) {
      v16bf a = load_a_frag(sPw, 64, 0, ks * 32);
      #pragma unroll
      for (int t = 0; t < 4; ++t) {
        v16bf bf = load_b_frag(sVt, 64, t * 16, ks * 32);
        Oacc[t] = __builtin_amdgcn_wmma_f32_16x16x32_bf16(
            false, a, false, bf, (short)0, Oacc[t], false, false);
      }
    }
  }

  #pragma unroll
  for (int j = 0; j < 8; ++j) {
    const int n_pos = q0 + wave * 16 + half * 8 + j;
    const float inv_l = 1.0f / l_i[j];
    #pragma unroll
    for (int t = 0; t < 4; ++t) {
      int d = t * 16 + l15;
      obf[((size_t)b * N_ + n_pos) * D_ + h * DH_ + d] = f2bf(Oacc[t][j] * inv_l);
    }
  }
}

extern "C" void kernel_launch(void* const* d_in, const int* in_sizes, int n_in,
                              void* d_out, int out_size, void* d_ws, size_t ws_size,
                              hipStream_t stream) {
  (void)in_sizes; (void)n_in; (void)out_size; (void)ws_size;
  const float* x          = (const float*)d_in[0];
  const float* w_qkv      = (const float*)d_in[1];
  const float* w_o        = (const float*)d_in[2];
  const float* log_scale  = (const float*)d_in[3];
  const float* log_lambda = (const float*)d_in[4];
  float* out = (float*)d_out;

  char* ws = (char*)d_ws;
  size_t off = 0;
  auto alloc = [&](size_t bytes) -> void* {
    void* p = ws + off;
    off += (bytes + 255) & ~(size_t)255;
    return p;
  };
  const int M = B_ * N_;                                  // 4096
  unsigned short* xbf    = (unsigned short*)alloc((size_t)M * D_ * 2);
  unsigned short* wqkvbf = (unsigned short*)alloc((size_t)D_ * 3 * D_ * 2);
  unsigned short* wobf   = (unsigned short*)alloc((size_t)D_ * D_ * 2);
  float* cosT  = (float*)alloc((size_t)H_ * N_ * KP_ * 4);
  float* sinT  = (float*)alloc((size_t)H_ * N_ * KP_ * 4);
  float* kbias = (float*)alloc((size_t)H_ * N_ * 4);
  float* qkvf  = (float*)alloc((size_t)M * 3 * D_ * 4);
  unsigned short* qbf = (unsigned short*)alloc((size_t)B_ * H_ * N_ * DH_ * 2);
  unsigned short* kbf = (unsigned short*)alloc((size_t)B_ * H_ * N_ * DH_ * 2);
  unsigned short* vbf = (unsigned short*)alloc((size_t)B_ * H_ * N_ * DH_ * 2);
  unsigned short* attnbf = (unsigned short*)alloc((size_t)M * D_ * 2);

  // 1. bf16 conversions
  k_f32_to_bf16<<<(M * D_ + 255) / 256, 256, 0, stream>>>(x, xbf, M * D_);
  k_f32_to_bf16<<<(D_ * 3 * D_ + 255) / 256, 256, 0, stream>>>(w_qkv, wqkvbf, D_ * 3 * D_);
  k_f32_to_bf16<<<(D_ * D_ + 255) / 256, 256, 0, stream>>>(w_o, wobf, D_ * D_);

  // 2. rotary tables + rank-1 sheet bias
  k_tables<<<(H_ * N_ + 255) / 256, 256, 0, stream>>>(log_scale, log_lambda, cosT, sinT, kbias);

  // 3. qkv = x @ w_qkv   (4096 x 512 x 1536)
  k_gemm_bf16<D_, 3 * D_, 3 * D_, D_>
      <<<dim3(3 * D_ / 64, M / 128), 256, 0, stream>>>(xbf, wqkvbf, qkvf);

  // 4. rotary + scatter to [B,H,N,dh] bf16
  k_rotary<<<(B_ * H_ * N_ * KP_) / 256, 256, 0, stream>>>(qkvf, cosT, sinT, qbf, kbf, vbf);

  // 5. causal flash attention with per-key sheet bias
  k_attn<<<dim3(N_ / 128, B_ * H_), 256, 0, stream>>>(qbf, kbf, vbf, kbias, attnbf);

  // 6. out = attn @ w_o   (4096 x 512 x 512), fp32 result
  k_gemm_bf16<D_, D_, D_, D_>
      <<<dim3(D_ / 64, M / 128), 256, 0, stream>>>(attnbf, wobf, out);
}
